// MyLSTM_22445499088922
// MI455X (gfx1250) — matrix operand order repaired and checked by reading, not compiled, not run on hardware
//
#include <hip/hip_runtime.h>
#include <hip/hip_bf16.h>

// ---------------- problem constants ----------------
#define kB    640
#define kT    160
#define kMEL  40
#define kXPAD 64
#define kHID  768
#define kG4   3072      // 4*kHID
#define kEMB  256
#define kAStride 1560   // LDS A-panel row stride in halves (>= xld+768, bank-staggered)

typedef _Float16 f16;
typedef __attribute__((ext_vector_type(16))) _Float16 v16h;
typedef __attribute__((ext_vector_type(8)))  _Float16 v8h;
typedef __attribute__((ext_vector_type(8)))  float    v8f;

#define WMMA_F16(a, b, c) \
    __builtin_amdgcn_wmma_f32_16x16x32_f16(false, (a), false, (b), (short)0, (c), false, false)

// CDNA5 async global->LDS copy path (guarded so compile can never break)
#if defined(__AMDGCN__) && __has_builtin(__builtin_amdgcn_global_load_async_to_lds_b128) && \
    __has_builtin(__builtin_amdgcn_s_wait_asynccnt)
#define ASYNC_LDS 1
typedef int v4i __attribute__((vector_size(16)));
typedef __attribute__((address_space(1))) v4i as1_v4i;   // global int4
typedef __attribute__((address_space(3))) v4i as3_v4i;   // LDS int4
#else
#define ASYNC_LDS 0
#endif

// ---- A fragment from LDS panel: 16x32 f16, per-ISA lane layout ----
// lane 0-15 : M=m0+lane,    a[0..7]=K(k0+0..7),  a[8..15]=K(k0+16..23)
// lane 16-31: M=m0+lane-16, a[0..7]=K(k0+8..15), a[8..15]=K(k0+24..31)
__device__ __forceinline__ v16h load_a_lds(const f16* As, int m0, int k0, int lane) {
    int m     = m0 + (lane & 15);
    int basek = (lane >> 4) * 8;
    const f16* p = As + m * kAStride + k0 + basek;
    v8h lo = *(const v8h*)(p);
    v8h hi = *(const v8h*)(p + 16);
    return __builtin_shufflevector(lo, hi, 0,1,2,3,4,5,6,7,8,9,10,11,12,13,14,15);
}

// ---- A fragment from global, row-major source (dense tail) ----
__device__ __forceinline__ v16h load_a_frag(const f16* __restrict__ Abase,
                                            long row_stride, int k0, int lane) {
    int m     = lane & 15;
    int basek = (lane >> 4) * 8;
    const f16* p = Abase + (long)m * row_stride + k0 + basek;
    v8h lo = *(const v8h*)(p);
    v8h hi = *(const v8h*)(p + 16);
    return __builtin_shufflevector(lo, hi, 0,1,2,3,4,5,6,7,8,9,10,11,12,13,14,15);
}

// ---- B fragment: 32x16 f16. B[k][n] = W[n][k], W row-major [N,K] ----
__device__ __forceinline__ v16h load_b_frag(const f16* __restrict__ W, long ldw,
                                            int n0, int k0, int lane) {
    int n    = lane & 15;
    int koff = (lane >> 4) * 8;
    const f16* p = W + (long)(n0 + n) * ldw + k0 + koff;
    v8h lo = *(const v8h*)(p);
    v8h hi = *(const v8h*)(p + 16);
    return __builtin_shufflevector(lo, hi, 0,1,2,3,4,5,6,7,8,9,10,11,12,13,14,15);
}

// ---------------- conversion / setup kernels ----------------
__global__ void cvt_f32_f16_kernel(const float* __restrict__ s, f16* __restrict__ d, int n) {
    int i = blockIdx.x * 256 + threadIdx.x;
    if (i < n) d[i] = (f16)s[i];
}

__global__ void cvt_pad_kernel(const float* __restrict__ s, f16* __restrict__ d,
                               int rows, int scols, int dcols) {
    int i = blockIdx.x * 256 + threadIdx.x;
    if (i >= rows * dcols) return;
    int r = i / dcols, c = i - r * dcols;
    d[i] = (c < scols) ? (f16)s[r * scols + c] : (f16)0.0f;
}

__global__ void bias_sum_kernel(const float* __restrict__ a, const float* __restrict__ b,
                                float* __restrict__ o, int n) {
    int i = blockIdx.x * 256 + threadIdx.x;
    if (i < n) o[i] = a[i] + b[i];
}

__global__ void zero_f32_kernel(float* __restrict__ p, int n) {
    int i = blockIdx.x * 256 + threadIdx.x;
    if (i < n) p[i] = 0.0f;
}

// ---------------- fused LSTM step ----------------
// grid (20, 12): blockIdx.x = batch tile (32 rows), blockIdx.y = hid tile (64 cols)
// 128 threads = 4 waves; wave w computes gate w's 32x64 tile (2 M-subtiles x 4 N-subtiles).
__global__ __launch_bounds__(128)
void lstm_step_kernel(const f16* __restrict__ xseq, int xld,          // [B,T,xld] f16
                      const f16* __restrict__ wih,                    // [3072,xld] f16
                      const f16* __restrict__ whh,                    // [3072,768] f16
                      const float* __restrict__ bias,                 // [3072] (b_ih+b_hh)
                      f16* __restrict__ hseq,                         // [B,T,768] f16 (out; h_{t-1} read)
                      float* __restrict__ c_state,                    // [B,768] f32
                      int t) {
    const int  btile = blockIdx.x;
    const int  hid0  = blockIdx.y * 64;
    const int  wave  = threadIdx.x >> 5;      // gate index 0..3 (i,f,g,o)
    const int  lane  = threadIdx.x & 31;
    const int  tid   = threadIdx.x;
    const int  gN0   = wave * kHID + hid0;    // gate column base in [0,3072)
    const long b0    = (long)btile * 32;

    // A panel [32][kAStride] halves; later reused as float gate buffer [4][32][68]
    __shared__ __align__(16) f16 smem[32 * kAStride];

    // ---- stage A panel: cols [0,xld) = x_t ; cols [xld, xld+768) = h_{t-1} (zeros at t=0) ----
    {
        const int xc8 = xld >> 3;                 // 8-half chunks per x row
        for (int ch = tid; ch < 32 * xc8; ch += 128) {
            int row = ch / xc8, c8 = ch - row * xc8;
            const f16* g = xseq + ((b0 + row) * kT + t) * xld + c8 * 8;
            f16* l = smem + row * kAStride + c8 * 8;
#if ASYNC_LDS
            __builtin_amdgcn_global_load_async_to_lds_b128((as1_v4i*)g, (as3_v4i*)l, 0, 0);
#else
            *(v8h*)l = *(const v8h*)g;
#endif
        }
        if (t > 0) {
            for (int ch = tid; ch < 32 * 96; ch += 128) {
                int row = ch / 96, c8 = ch - row * 96;
                const f16* g = hseq + ((b0 + row) * kT + (t - 1)) * kHID + c8 * 8;
                f16* l = smem + row * kAStride + xld + c8 * 8;
#if ASYNC_LDS
                __builtin_amdgcn_global_load_async_to_lds_b128((as1_v4i*)g, (as3_v4i*)l, 0, 0);
#else
                *(v8h*)l = *(const v8h*)g;
#endif
            }
        } else {
            v8h z = {};
            for (int ch = tid; ch < 32 * 96; ch += 128) {
                int row = ch / 96, c8 = ch - row * 96;
                *(v8h*)(smem + row * kAStride + xld + c8 * 8) = z;
            }
        }
    }
#if ASYNC_LDS
    __builtin_amdgcn_s_wait_asynccnt(0);
#endif
    __syncthreads();

    // ---- GEMM: gates[32 x 64] for this wave's gate, K = xld (+768 recurrent) ----
    v8f acc[2][4] = {};

    #pragma unroll 2
    for (int k0 = 0; k0 < xld; k0 += 32) {
        v16h bf0 = load_b_frag(wih, xld, gN0 +  0, k0, lane);
        v16h bf1 = load_b_frag(wih, xld, gN0 + 16, k0, lane);
        v16h bf2 = load_b_frag(wih, xld, gN0 + 32, k0, lane);
        v16h bf3 = load_b_frag(wih, xld, gN0 + 48, k0, lane);
        v16h a0  = load_a_lds(smem,  0, k0, lane);
        v16h a1  = load_a_lds(smem, 16, k0, lane);
        acc[0][0] = WMMA_F16(a0, bf0, acc[0][0]);
        acc[1][0] = WMMA_F16(a1, bf0, acc[1][0]);
        acc[0][1] = WMMA_F16(a0, bf1, acc[0][1]);
        acc[1][1] = WMMA_F16(a1, bf1, acc[1][1]);
        acc[0][2] = WMMA_F16(a0, bf2, acc[0][2]);
        acc[1][2] = WMMA_F16(a1, bf2, acc[1][2]);
        acc[0][3] = WMMA_F16(a0, bf3, acc[0][3]);
        acc[1][3] = WMMA_F16(a1, bf3, acc[1][3]);
    }
    #pragma unroll 2
    for (int k0 = 0; k0 < kHID; k0 += 32) {
        v16h bf0 = load_b_frag(whh, kHID, gN0 +  0, k0, lane);
        v16h bf1 = load_b_frag(whh, kHID, gN0 + 16, k0, lane);
        v16h bf2 = load_b_frag(whh, kHID, gN0 + 32, k0, lane);
        v16h bf3 = load_b_frag(whh, kHID, gN0 + 48, k0, lane);
        v16h a0  = load_a_lds(smem,  0, xld + k0, lane);
        v16h a1  = load_a_lds(smem, 16, xld + k0, lane);
        acc[0][0] = WMMA_F16(a0, bf0, acc[0][0]);
        acc[1][0] = WMMA_F16(a1, bf0, acc[1][0]);
        acc[0][1] = WMMA_F16(a0, bf1, acc[0][1]);
        acc[1][1] = WMMA_F16(a1, bf1, acc[1][1]);
        acc[0][2] = WMMA_F16(a0, bf2, acc[0][2]);
        acc[1][2] = WMMA_F16(a1, bf2, acc[1][2]);
        acc[0][3] = WMMA_F16(a0, bf3, acc[0][3]);
        acc[1][3] = WMMA_F16(a1, bf3, acc[1][3]);
    }

    // ---- reuse smem as gate buffer gl[4][32][68] floats (34.8 KB) ----
    __syncthreads();                              // everyone done reading A panel
    float* gl = (float*)smem;
    {
        const int mb = (lane >> 4) * 8;
        const int nn = lane & 15;
        #pragma unroll
        for (int ms = 0; ms < 2; ++ms)
            #pragma unroll
            for (int j = 0; j < 4; ++j)
                #pragma unroll
                for (int r = 0; r < 8; ++r)
                    gl[((wave * 32) + (ms * 16 + mb + r)) * 68 + j * 16 + nn] = acc[ms][j][r];
    }
    __syncthreads();

    // ---- elementwise cell update: 2048 elements / 128 threads = 16 each ----
    #pragma unroll
    for (int e = 0; e < 16; ++e) {
        int flat = tid + 128 * e;
        int m = flat >> 6;
        int n = flat & 63;
        int col = hid0 + n;
        float gi = gl[(0 * 32 + m) * 68 + n] + bias[0 * kHID + col];
        float gf = gl[(1 * 32 + m) * 68 + n] + bias[1 * kHID + col];
        float gg = gl[(2 * 32 + m) * 68 + n] + bias[2 * kHID + col];
        float go = gl[(3 * 32 + m) * 68 + n] + bias[3 * kHID + col];
        float iv = 1.0f / (1.0f + __expf(-gi));
        float fv = 1.0f / (1.0f + __expf(-gf));
        float gv = tanhf(gg);
        float ov = 1.0f / (1.0f + __expf(-go));
        long  b    = b0 + m;
        long  cidx = b * kHID + col;
        float c = fv * c_state[cidx] + iv * gv;
        c_state[cidx] = c;
        float h = ov * tanhf(c);
        hseq[(b * kT + t) * kHID + col] = (f16)h;
    }
}

// ---------------- final dense (WMMA) ----------------
// grid (40, 4), 128 threads; wave w -> n0 = blockIdx.y*64 + w*16
__global__ __launch_bounds__(128)
void dense_kernel(const f16* __restrict__ hseq,     // [B,T,768]; reads t = T-1
                  const f16* __restrict__ wd,       // [256,768] f16
                  const float* __restrict__ bd,     // [256]
                  float* __restrict__ y) {          // [B,256]
    const int btile = blockIdx.x;
    const int lane  = threadIdx.x & 31;
    const int n0    = blockIdx.y * 64 + (threadIdx.x >> 5) * 16;
    const f16* A  = hseq + ((long)btile * 16 * kT + (kT - 1)) * kHID;
    const long rs = (long)kT * kHID;
    v8f acc = {};
    #pragma unroll 4
    for (int k0 = 0; k0 < kHID; k0 += 32) {
        v16h a = load_a_frag(A, rs, k0, lane);
        v16h b = load_b_frag(wd, kHID, n0, k0, lane);
        acc = WMMA_F16(a, b, acc);
    }
    const int mb = (lane >> 4) * 8;
    const int nn = lane & 15;
    #pragma unroll
    for (int r = 0; r < 8; ++r) {
        int n = n0 + nn;
        y[(long)(btile * 16 + mb + r) * kEMB + n] = acc[r] + bd[n];
    }
}

// ---------------- L2 normalize ----------------
__global__ __launch_bounds__(256)
void norm_kernel(const float* __restrict__ y, float* __restrict__ out) {
    __shared__ float red[256];
    const int b = blockIdx.x, tid = threadIdx.x;
    float v = y[(long)b * kEMB + tid];
    red[tid] = v * v;
    __syncthreads();
    #pragma unroll
    for (int s = 128; s > 0; s >>= 1) {
        if (tid < s) red[tid] += red[tid + s];
        __syncthreads();
    }
    float inv = rsqrtf(red[0]);
    out[(long)b * kEMB + tid] = v * inv;
}

// ---------------- host orchestration ----------------
extern "C" void kernel_launch(void* const* d_in, const int* in_sizes, int n_in,
                              void* d_out, int out_size, void* d_ws, size_t ws_size,
                              hipStream_t stream) {
    (void)in_sizes; (void)n_in; (void)out_size; (void)ws_size;
    const float* x     = (const float*)d_in[0];
    const float* wih_f[3] = { (const float*)d_in[1], (const float*)d_in[5], (const float*)d_in[9]  };
    const float* whh_f[3] = { (const float*)d_in[2], (const float*)d_in[6], (const float*)d_in[10] };
    const float* bih_f[3] = { (const float*)d_in[3], (const float*)d_in[7], (const float*)d_in[11] };
    const float* bhh_f[3] = { (const float*)d_in[4], (const float*)d_in[8], (const float*)d_in[12] };
    const float* wd_f  = (const float*)d_in[13];
    const float* bd_f  = (const float*)d_in[14];

    // workspace carve-out (256B aligned regions)
    char* cur = (char*)d_ws;
    auto carve = [&](size_t bytes) { char* p = cur; cur += (bytes + 255) & ~(size_t)255; return p; };
    f16*   X16   = (f16*)  carve((size_t)kB * kT * kXPAD * 2);     // padded f16 input
    f16*   seqA  = (f16*)  carve((size_t)kB * kT * kHID * 2);      // layer h sequences (ping)
    f16*   seqB  = (f16*)  carve((size_t)kB * kT * kHID * 2);      // (pong)
    f16*   wih16[3];
    wih16[0] = (f16*) carve((size_t)kG4 * kXPAD * 2);
    wih16[1] = (f16*) carve((size_t)kG4 * kHID * 2);
    wih16[2] = (f16*) carve((size_t)kG4 * kHID * 2);
    f16*   whh16[3];
    for (int l = 0; l < 3; ++l) whh16[l] = (f16*) carve((size_t)kG4 * kHID * 2);
    f16*   wd16  = (f16*)  carve((size_t)kEMB * kHID * 2);
    float* bias[3];
    for (int l = 0; l < 3; ++l) bias[l] = (float*) carve((size_t)kG4 * 4);
    float* cst   = (float*) carve((size_t)kB * kHID * 4);
    float* y     = (float*) carve((size_t)kB * kEMB * 4);

    auto blocks = [](int n) { return (n + 255) / 256; };

    // --- setup / conversion ---
    cvt_pad_kernel<<<blocks(kB * kT * kXPAD), 256, 0, stream>>>(x, X16, kB * kT, kMEL, kXPAD);
    cvt_pad_kernel<<<blocks(kG4 * kXPAD), 256, 0, stream>>>(wih_f[0], wih16[0], kG4, kMEL, kXPAD);
    cvt_f32_f16_kernel<<<blocks(kG4 * kHID), 256, 0, stream>>>(wih_f[1], wih16[1], kG4 * kHID);
    cvt_f32_f16_kernel<<<blocks(kG4 * kHID), 256, 0, stream>>>(wih_f[2], wih16[2], kG4 * kHID);
    for (int l = 0; l < 3; ++l) {
        cvt_f32_f16_kernel<<<blocks(kG4 * kHID), 256, 0, stream>>>(whh_f[l], whh16[l], kG4 * kHID);
        bias_sum_kernel<<<blocks(kG4), 256, 0, stream>>>(bih_f[l], bhh_f[l], bias[l], kG4);
    }
    cvt_f32_f16_kernel<<<blocks(kEMB * kHID), 256, 0, stream>>>(wd_f, wd16, kEMB * kHID);

    // --- LSTM layers (sequential over t; launch boundary = inter-step sync) ---
    const dim3 step_grid(kB / 32, kHID / 64);   // 20 x 12
    for (int l = 0; l < 3; ++l) {
        zero_f32_kernel<<<blocks(kB * kHID), 256, 0, stream>>>(cst, kB * kHID);
        const f16* xin;  int xld;  f16* hout;
        if (l == 0)      { xin = X16;  xld = kXPAD; hout = seqA; }
        else if (l == 1) { xin = seqA; xld = kHID;  hout = seqB; }
        else             { xin = seqB; xld = kHID;  hout = seqA; }
        for (int t = 0; t < kT; ++t) {
            lstm_step_kernel<<<step_grid, 128, 0, stream>>>(
                xin, xld, wih16[l], whh16[l], bias[l], hout, cst, t);
        }
    }

    // --- dense + L2 normalize ---
    dense_kernel<<<dim3(kB / 16, kEMB / 64), 128, 0, stream>>>(seqA, wd16, bd_f, y);
    norm_kernel<<<kB, 256, 0, stream>>>(y, (float*)d_out);
}
